// AttentionBlock_72438918414678
// MI455X (gfx1250) — matrix operand order repaired and checked by reading, not compiled, not run on hardware
//
#include <hip/hip_runtime.h>
#include <stdint.h>

// ---------------------------------------------------------------------------
// AttentionBlock for MI455X (gfx1250, wave32, WMMA bf16)
// B=16, C=512, H=W=32 (S=1024 spatial), GROUPS=32
// ---------------------------------------------------------------------------

#define B_      16
#define C_      512
#define S_      1024
#define GROUPS_ 32
#define GSIZE_  16384   // (C_/GROUPS_) * S_  contiguous elems per (b,group)

typedef __attribute__((ext_vector_type(16))) __bf16 v16bf;
typedef __attribute__((ext_vector_type(8)))  float  v8f;

union FragBF { v16bf v; uint32_t u[8]; };

__device__ __forceinline__ uint16_t f2bf(float f) {
  uint32_t u = __float_as_uint(f);
  u += 0x7FFFu + ((u >> 16) & 1u);      // round-to-nearest-even
  return (uint16_t)(u >> 16);
}
__device__ __forceinline__ float bf2f_lo(uint32_t u) {
  return __uint_as_float(u << 16);
}
__device__ __forceinline__ float bf2f_hi(uint32_t u) {
  return __uint_as_float(u & 0xFFFF0000u);
}
__device__ __forceinline__ uint32_t pack_bf2(float lo, float hi) {
  return (uint32_t)f2bf(lo) | ((uint32_t)f2bf(hi) << 16);
}
__device__ __forceinline__ v8f zero8() {
  v8f z;
#pragma unroll
  for (int i = 0; i < 8; ++i) z[i] = 0.0f;
  return z;
}

// ---------------------------------------------------------------------------
// Kernel 1: GroupNorm statistics. One block per (b, group); group data is a
// contiguous 16384-float chunk. 512 blocks x 256 threads, float4 loads.
// ---------------------------------------------------------------------------
__global__ void __launch_bounds__(256) gn_stats_kernel(
    const float* __restrict__ x, float* __restrict__ mu, float* __restrict__ rstd) {
  const int bg = blockIdx.x;
  const float4* p4 = (const float4*)(x + (size_t)bg * GSIZE_);
  float s = 0.f, ss = 0.f;
  for (int i = threadIdx.x; i < GSIZE_ / 4; i += 256) {
    float4 v = p4[i];
    s  += v.x + v.y + v.z + v.w;
    ss += v.x * v.x + v.y * v.y + v.z * v.z + v.w * v.w;
  }
#pragma unroll
  for (int off = 16; off >= 1; off >>= 1) {
    s  += __shfl_xor(s,  off, 32);
    ss += __shfl_xor(ss, off, 32);
  }
  __shared__ float ps[8], pss[8];
  const int wid = threadIdx.x >> 5, lane = threadIdx.x & 31;
  if (lane == 0) { ps[wid] = s; pss[wid] = ss; }
  __syncthreads();
  if (threadIdx.x == 0) {
    float ts = 0.f, tss = 0.f;
#pragma unroll
    for (int i = 0; i < 8; ++i) { ts += ps[i]; tss += pss[i]; }
    float m   = ts  * (1.0f / GSIZE_);
    float var = tss * (1.0f / GSIZE_) - m * m;
    mu[bg]   = m;
    rstd[bg] = rsqrtf(var + 1e-5f);
  }
}

// ---------------------------------------------------------------------------
// Kernel 2: normalize + transpose + bf16 convert.
// x[b][c][s] (f32) -> hbf[b][s][c] (bf16).  Tile 32c x 128s via LDS; float4
// global loads, packed-dword bf16 stores (both coalesced).
// ---------------------------------------------------------------------------
__global__ void __launch_bounds__(256) gn_norm_kernel(
    const float* __restrict__ x, const float* __restrict__ mu,
    const float* __restrict__ rstd, uint16_t* __restrict__ hbf) {
  __shared__ float til[32][132];
  const int s0 = blockIdx.x * 128, c0 = blockIdx.y * 32, b = blockIdx.z;
  for (int i = threadIdx.x; i < 1024; i += 256) {
    int cl = i >> 5, sl4 = (i & 31) * 4;
    int c  = c0 + cl;
    int bg = b * GROUPS_ + (c >> 4);
    float m = mu[bg], r = rstd[bg];
    float4 v = *(const float4*)(x + ((size_t)b * C_ + c) * S_ + s0 + sl4);
    til[cl][sl4 + 0] = (v.x - m) * r;
    til[cl][sl4 + 1] = (v.y - m) * r;
    til[cl][sl4 + 2] = (v.z - m) * r;
    til[cl][sl4 + 3] = (v.w - m) * r;
  }
  __syncthreads();
  uint32_t* h32 = (uint32_t*)hbf;
  for (int i = threadIdx.x; i < 2048; i += 256) {
    int sl = i >> 4, c2 = i & 15;
    h32[(((size_t)b * S_ + s0 + sl) * C_ + c0) / 2 + c2] =
        pack_bf2(til[2 * c2][sl], til[2 * c2 + 1][sl]);
  }
}

// ---------------------------------------------------------------------------
// Kernel 3: f32 -> bf16 convert (weights), packed dword stores
// ---------------------------------------------------------------------------
__global__ void cvt_bf16_kernel(const float* __restrict__ src,
                                uint16_t* __restrict__ dst, int n2) {
  int i = blockIdx.x * 256 + threadIdx.x;   // n2 = n/2 dwords
  if (i < n2) {
    float2 v = *(const float2*)(src + 2 * i);
    ((uint32_t*)dst)[i] = pack_bf2(v.x, v.y);
  }
}

// ---------------------------------------------------------------------------
// Kernel 4: NIN GEMM  Out[s][o] = sum_c A[s][c] * W[c][o] + bias[o]
//   MODE 0: A = hbf, store bf16 spatially transposed (s' = (s%32)*32 + s/32)
//   MODE 1: A = hbf, store bf16 natural layout (V)
//   MODE 2: A = Vbf row-scaled by D2[b][s]; out f32 = x + acc + bias (final)
// 64x64 tile / workgroup, 4 waves, each wave 32x32 via 4 WMMA accumulators.
// Fragment layouts follow CDNA5 ISA 7.12.2 (bf16 A 16x32, B as K-pairs).
// All LDS traffic is dword-granular; MODE is compile-time (no hot branches).
// ---------------------------------------------------------------------------
template <int MODE>
__global__ void __launch_bounds__(128) nin_gemm_kernel(
    const uint16_t* __restrict__ Abf,   // [B][S][C] bf16
    const uint16_t* __restrict__ Wbf,   // [C][C] bf16
    const float*    __restrict__ bias,  // [C]
    const float*    __restrict__ D2,    // [B][S]  (MODE 2 only)
    const float*    __restrict__ xres,  // [B][C][S] (MODE 2 only)
    uint16_t* __restrict__ outbf,       // MODE 0/1
    float*    __restrict__ outf) {      // MODE 2
  __shared__ uint32_t As32[64 * 17];    // [row s][k-pair], stride 17 dwords
  __shared__ uint32_t Bs32[64 * 17];    // [col n][k-pair], stride 17 dwords
  const int s0 = blockIdx.x * 64, o0 = blockIdx.y * 64, b = blockIdx.z;
  const int tid = threadIdx.x, lane = tid & 31, wid = tid >> 5;
  const int wm = (wid & 1) * 32, wn = (wid >> 1) * 32;
  const uint32_t* A32 = (const uint32_t*)(Abf + (size_t)b * S_ * C_);
  const uint32_t* W32 = (const uint32_t*)Wbf;
  const float*    D2b = (MODE == 2) ? (D2 + (size_t)b * S_) : nullptr;

  v8f acc[2][2];
  acc[0][0] = zero8(); acc[0][1] = zero8();
  acc[1][0] = zero8(); acc[1][1] = zero8();

  for (int k0 = 0; k0 < C_; k0 += 32) {
    // ---- Stage A tile: 64 rows (s) x 16 k-pairs, coalesced dword loads ----
    for (int i = tid; i < 1024; i += 128) {
      int r = i >> 4, kd = i & 15;
      uint32_t u = A32[(size_t)(s0 + r) * (C_ / 2) + (k0 >> 1) + kd];
      if constexpr (MODE == 2) {
        float d = D2b[s0 + r];
        u = pack_bf2(bf2f_lo(u) * d, bf2f_hi(u) * d);
      }
      As32[r * 17 + kd] = u;
    }
    // ---- Stage B tile: 16 k-pairs x 64 n, repacked n-major ----
    for (int i = tid; i < 512; i += 128) {
      int kp = i >> 5, n2 = i & 31;     // k-pair 0..15, n-pair 0..31
      uint32_t a = W32[(size_t)(k0 + 2 * kp)     * (C_ / 2) + (o0 >> 1) + n2];
      uint32_t bq2 = W32[(size_t)(k0 + 2 * kp + 1) * (C_ / 2) + (o0 >> 1) + n2];
      Bs32[(2 * n2)     * 17 + kp] = (a & 0xFFFFu) | (bq2 << 16);
      Bs32[(2 * n2 + 1) * 17 + kp] = (a >> 16) | (bq2 & 0xFFFF0000u);
    }
    if (k0 + 32 < C_) {  // prefetch next k-tile (global_prefetch_b8)
      __builtin_prefetch(A32 + (size_t)(s0 + (tid >> 1)) * (C_ / 2) + ((k0 + 32) >> 1), 0, 1);
      __builtin_prefetch(W32 + (size_t)(k0 + 32 + (tid & 31)) * (C_ / 2) + (o0 >> 1), 0, 1);
    }
    __syncthreads();

    FragBF af[2], bfB[2];
    const int ka = (lane >= 16) ? 4 : 0;
    const int kb = (lane >= 16) ? 8 : 0;
#pragma unroll
    for (int mi = 0; mi < 2; ++mi) {
      int m = wm + mi * 16 + (lane & 15);
#pragma unroll
      for (int j = 0; j < 8; ++j)       // ISA A-layout (16x32 bf16)
        af[mi].u[j] = As32[m * 17 + ((j & 4) ? 8 : 0) + ka + (j & 3)];
    }
#pragma unroll
    for (int ni = 0; ni < 2; ++ni) {
      int n = wn + ni * 16 + (lane & 15);
#pragma unroll
      for (int r = 0; r < 8; ++r)       // ISA B-layout (K-pairs per dword)
        bfB[ni].u[r] = Bs32[n * 17 + ((r & 4) ? 8 : 0) + ka + (r & 3)];
    }
#pragma unroll
    for (int mi = 0; mi < 2; ++mi)
#pragma unroll
      for (int ni = 0; ni < 2; ++ni)
        acc[mi][ni] = __builtin_amdgcn_wmma_f32_16x16x32_bf16(
            false, af[mi].v, false, bfB[ni].v, (short)0, acc[mi][ni],
            false, false);
    __syncthreads();
  }

  // Epilogue (ISA C/D layout: reg r -> M = r + 8*(lane>=16), N = lane&15)
  const float bias0 = bias[o0 + wn +      (lane & 15)];
  const float bias1 = bias[o0 + wn + 16 + (lane & 15)];
#pragma unroll
  for (int mi = 0; mi < 2; ++mi) {
#pragma unroll
    for (int ni = 0; ni < 2; ++ni) {
      const int n = wn + ni * 16 + (lane & 15);
      const int o = o0 + n;
      const float bo = ni ? bias1 : bias0;
#pragma unroll
      for (int r = 0; r < 8; ++r) {
        int m = wm + mi * 16 + r + ((lane >= 16) ? 8 : 0);
        int s = s0 + m;
        float v = acc[mi][ni][r] + bo;
        if constexpr (MODE == 0) {
          int sp = ((s & 31) << 5) | (s >> 5);   // spatial transpose for Q/K
          outbf[(size_t)b * S_ * C_ + (size_t)sp * C_ + o] = f2bf(v);
        } else if constexpr (MODE == 1) {
          outbf[(size_t)b * S_ * C_ + (size_t)s * C_ + o] = f2bf(v);
        } else {
          size_t idx = (size_t)b * C_ * S_ + (size_t)o * S_ + s;
          outf[idx] = xres[idx] + v;             // residual add
        }
      }
    }
  }
}

// ---------------------------------------------------------------------------
// Kernel 5: attention diagonal-of-block-softmax.
// Workgroup per (b, a=w): shared Q tile (rows a*32..a*32+31 of Qt) + per-wave
// private K tile. Wave computes S-block(a, W) = Qt_tile . Kt_tileT (32x32),
// then block max / sum(exp) / entry (h=a, H=W) ->
//   D2[b][a*32+W] = exp(scale*S[a][W]-m) / sum exp(scale*S - m)
// Dynamic LDS: (1 + 8) * 32 rows * 260 dwords = 299,520 B (< 320 KB WGP).
// Staging uses b128 (rows are 1040 B -> 16 B aligned).
// ---------------------------------------------------------------------------
__global__ void __launch_bounds__(256) attn_diag_kernel(
    const uint16_t* __restrict__ Qt,  // [B][S][C] bf16, s' = w*32+h
    const uint16_t* __restrict__ Kt,  // [B][S][C] bf16, s' = W*32+H
    float* __restrict__ D2) {         // [B][S]
  extern __shared__ uint32_t sm[];
  const int a = blockIdx.x, b = blockIdx.y;
  const int tid = threadIdx.x, lane = tid & 31, wid = tid >> 5;
  uint32_t* Qs = sm;                            // 32 rows * 260 dwords
  uint32_t* Ks = sm + 32 * 260 * (1 + wid);     // per-wave region

  const uint4* Qg4 =
      (const uint4*)(Qt + (size_t)b * S_ * C_ + (size_t)a * 32 * C_);
  for (int i = tid; i < 2048; i += 256)         // 32 rows * 64 uint4
    ((uint4*)Qs)[(i >> 6) * 65 + (i & 63)] = Qg4[i];
  __syncthreads();

  const float scale = 0.044194173824159216f;    // 512^-0.5

  for (int it = 0; it < 4; ++it) {
    const int W = wid + it * 8;
    const uint4* Kg4 =
        (const uint4*)(Kt + (size_t)b * S_ * C_ + (size_t)W * 32 * C_);
    for (int i = lane; i < 2048; i += 32)
      ((uint4*)Ks)[(i >> 6) * 65 + (i & 63)] = Kg4[i];

    v8f acc[2][2];
    acc[0][0] = zero8(); acc[0][1] = zero8();
    acc[1][0] = zero8(); acc[1][1] = zero8();

    const int ka = (lane >= 16) ? 4 : 0;
    const int kb = (lane >= 16) ? 8 : 0;
    for (int k0 = 0; k0 < 256; k0 += 16) {      // dword offsets, 32 bf16/step
      FragBF af[2], bfB[2];
#pragma unroll
      for (int mi = 0; mi < 2; ++mi) {
        int m = mi * 16 + (lane & 15);
#pragma unroll
        for (int j = 0; j < 8; ++j)
          af[mi].u[j] = Qs[m * 260 + k0 + ((j & 4) ? 8 : 0) + ka + (j & 3)];
      }
#pragma unroll
      for (int ni = 0; ni < 2; ++ni) {
        int n = ni * 16 + (lane & 15);
#pragma unroll
        for (int r = 0; r < 8; ++r)
          bfB[ni].u[r] = Ks[n * 260 + k0 + ((r & 4) ? 8 : 0) + ka + (r & 3)];
      }
#pragma unroll
      for (int mi = 0; mi < 2; ++mi)
#pragma unroll
        for (int ni = 0; ni < 2; ++ni)
          acc[mi][ni] = __builtin_amdgcn_wmma_f32_16x16x32_bf16(
              false, af[mi].v, false, bfB[ni].v, (short)0, acc[mi][ni],
              false, false);
    }

    // Block softmax stats over full 32x32 tile; grab entry (m=a, n=W).
    float vmax = -3.0e38f, tcand = -3.0e38f;
#pragma unroll
    for (int mi = 0; mi < 2; ++mi)
#pragma unroll
      for (int ni = 0; ni < 2; ++ni)
#pragma unroll
        for (int r = 0; r < 8; ++r) {
          int m = mi * 16 + r + ((lane >= 16) ? 8 : 0);
          int n = ni * 16 + (lane & 15);
          float v = acc[mi][ni][r] * scale;
          vmax = fmaxf(vmax, v);
          if (m == a && n == W) tcand = v;
        }
#pragma unroll
    for (int off = 16; off >= 1; off >>= 1) {
      vmax  = fmaxf(vmax,  __shfl_xor(vmax,  off, 32));
      tcand = fmaxf(tcand, __shfl_xor(tcand, off, 32));
    }
    float sume = 0.f;
#pragma unroll
    for (int mi = 0; mi < 2; ++mi)
#pragma unroll
      for (int ni = 0; ni < 2; ++ni)
#pragma unroll
        for (int r = 0; r < 8; ++r)
          sume += __expf(acc[mi][ni][r] * scale - vmax);
#pragma unroll
    for (int off = 16; off >= 1; off >>= 1)
      sume += __shfl_xor(sume, off, 32);

    if (lane == 0)
      D2[(size_t)b * S_ + a * 32 + W] = __expf(tcand - vmax) / sume;
  }
}

// ---------------------------------------------------------------------------
// Host-side launch
// ---------------------------------------------------------------------------
extern "C" void kernel_launch(void* const* d_in, const int* in_sizes, int n_in,
                              void* d_out, int out_size, void* d_ws, size_t ws_size,
                              hipStream_t stream) {
  (void)in_sizes; (void)n_in; (void)out_size; (void)ws_size;
  const float* x  = (const float*)d_in[0];
  const float* Wq = (const float*)d_in[1];
  const float* bq = (const float*)d_in[2];
  const float* Wk = (const float*)d_in[3];
  const float* bk = (const float*)d_in[4];
  const float* Wv = (const float*)d_in[5];
  const float* bv = (const float*)d_in[6];
  const float* Wn = (const float*)d_in[7];
  const float* bn = (const float*)d_in[8];
  float* out = (float*)d_out;
  char* ws = (char*)d_ws;

  constexpr size_t MU_OFF   = 0;
  constexpr size_t RSTD_OFF = 2048;
  constexpr size_t D2_OFF   = 4096;
  constexpr size_t HBF_OFF  = D2_OFF + (size_t)B_ * S_ * 4;       // 69,632
  constexpr size_t HBF_SZ   = (size_t)B_ * S_ * C_ * 2;           // 16 MB
  constexpr size_t W_SZ     = (size_t)C_ * C_ * 2;                // 512 KB
  constexpr size_t WQ_OFF   = HBF_OFF + HBF_SZ;
  constexpr size_t WK_OFF   = WQ_OFF + W_SZ;
  constexpr size_t WV_OFF   = WK_OFF + W_SZ;
  constexpr size_t WN_OFF   = WV_OFF + W_SZ;
  constexpr size_t QT_OFF   = WN_OFF + W_SZ;
  constexpr size_t KT_OFF   = QT_OFF + HBF_SZ;
  constexpr size_t VB_OFF   = KT_OFF + HBF_SZ;                    // end ~69 MB

  float*    muP   = (float*)(ws + MU_OFF);
  float*    rstdP = (float*)(ws + RSTD_OFF);
  float*    d2P   = (float*)(ws + D2_OFF);
  uint16_t* hbfP  = (uint16_t*)(ws + HBF_OFF);
  uint16_t* wqP   = (uint16_t*)(ws + WQ_OFF);
  uint16_t* wkP   = (uint16_t*)(ws + WK_OFF);
  uint16_t* wvP   = (uint16_t*)(ws + WV_OFF);
  uint16_t* wnP   = (uint16_t*)(ws + WN_OFF);
  uint16_t* qtP   = (uint16_t*)(ws + QT_OFF);
  uint16_t* ktP   = (uint16_t*)(ws + KT_OFF);
  uint16_t* vbP   = (uint16_t*)(ws + VB_OFF);

  // 1) GroupNorm
  gn_stats_kernel<<<B_ * GROUPS_, 256, 0, stream>>>(x, muP, rstdP);
  gn_norm_kernel<<<dim3(S_ / 128, C_ / 32, B_), 256, 0, stream>>>(x, muP, rstdP, hbfP);

  // 2) Weight conversion to bf16 (dword-packed)
  const int wN2 = (C_ * C_) / 2;
  cvt_bf16_kernel<<<wN2 / 256, 256, 0, stream>>>(Wq, wqP, wN2);
  cvt_bf16_kernel<<<wN2 / 256, 256, 0, stream>>>(Wk, wkP, wN2);
  cvt_bf16_kernel<<<wN2 / 256, 256, 0, stream>>>(Wv, wvP, wN2);
  cvt_bf16_kernel<<<wN2 / 256, 256, 0, stream>>>(Wn, wnP, wN2);

  // 3) Q/K/V projections (WMMA bf16)
  dim3 ggrid(S_ / 64, C_ / 64, B_);
  nin_gemm_kernel<0><<<ggrid, 128, 0, stream>>>(hbfP, wqP, bq, nullptr, nullptr, qtP, nullptr);
  nin_gemm_kernel<0><<<ggrid, 128, 0, stream>>>(hbfP, wkP, bk, nullptr, nullptr, ktP, nullptr);
  nin_gemm_kernel<1><<<ggrid, 128, 0, stream>>>(hbfP, wvP, bv, nullptr, nullptr, vbP, nullptr);

  // 4) Block-softmax diagonal (WMMA bf16 + wave32 reductions)
  size_t smem = (size_t)(1 + 8) * 32 * 260 * 4;   // 299,520 B dynamic LDS
  attn_diag_kernel<<<dim3(32, B_), 256, smem, stream>>>(qtP, ktP, d2P);

  // 5) Final NIN with fused diag scaling + residual -> d_out (f32)
  nin_gemm_kernel<2><<<ggrid, 128, 0, stream>>>(vbP, wnP, bn, d2P, x, nullptr, out);
}
